// MultiHeadSeqAttention_78091095376384
// MI455X (gfx1250) — matrix operand order repaired and checked by reading, not compiled, not run on hardware
//
#include <hip/hip_runtime.h>

// MI455X / gfx1250: wave32, WMMA f32_16x16x32_f16 path.
// Linear attention identity: (q k^T) v == q (k^T v)  -> per-head 64x64 inner matrix.
// Big GEMMs: block 256 thr = 8 waves, each wave 32x64 tile; B strip staged in
// double-buffered LDS (async-to-LDS when available), A from global (L2-resident).

typedef __attribute__((ext_vector_type(16))) _Float16 v16h;
typedef __attribute__((ext_vector_type(8)))  _Float16 v8h;
typedef __attribute__((ext_vector_type(8)))  float    v8f;
typedef __attribute__((ext_vector_type(4)))  int      i4;

// exact parameter types for __builtin_amdgcn_global_load_async_to_lds_b128:
// (i4 addrspace(1)*, i4 addrspace(3)*, imm offset, imm cpol)
typedef __attribute__((address_space(1))) i4 gl_i4;
typedef __attribute__((address_space(3))) i4 lds_i4;

#define HDIM 1024
#define MSEQ 1024
#define BSZ  8
#define BM   (BSZ * MSEQ)
#define NH   16
#define DH   64

#if __has_builtin(__builtin_amdgcn_global_load_async_to_lds_b128)
#define ASYNC_LDS 1
#if __has_builtin(__builtin_amdgcn_s_wait_asynccnt)
#define WAIT_ASYNC() __builtin_amdgcn_s_wait_asynccnt(0)
#else
#define WAIT_ASYNC() asm volatile("s_wait_asynccnt 0x0" ::: "memory")
#endif
#else
#define ASYNC_LDS 0
#endif

// ---------- helpers ----------

__device__ __forceinline__ v16h load16h(const _Float16* lo, const _Float16* hi) {
    v8h a = *(const v8h*)lo;
    v8h b = *(const v8h*)hi;
    return __builtin_shufflevector(a, b, 0,1,2,3,4,5,6,7,8,9,10,11,12,13,14,15);
}

__device__ __forceinline__ v8f wmma_f16(v16h a, v16h b, v8f c) {
    return __builtin_amdgcn_wmma_f32_16x16x32_f16(
        false, a, false, b, (short)0, c, false, false);
}

// ---------- fp32 -> fp16 convert ----------

__global__ void f32_to_f16_kernel(const float* __restrict__ src,
                                  _Float16* __restrict__ dst, int n) {
    int i = blockIdx.x * blockDim.x + threadIdx.x;
    int stride = gridDim.x * blockDim.x;
    for (; i < n; i += stride) dst[i] = (_Float16)src[i];
}

// ---------- unified big GEMM: P = A @ W^T ----------
// A:[BM,HDIM] f16 row-major; W:[HDIM,HDIM] f16 row-major (x@W^T => W rows are
// exactly the WMMA B-operand contiguous runs).
// mode 0: outPlain f16 [BM,HDIM]   (Q)
// mode 1: outTr f16 per-head transposed [B*NH][DH][MSEQ]   (Kt / Vt)
// mode 2: outF32 [BM,HDIM]   (final output)
__global__ void __launch_bounds__(256)
gemm_wmma_kernel(const _Float16* __restrict__ A,
                 const _Float16* __restrict__ W,
                 _Float16* __restrict__ outPlain,
                 _Float16* __restrict__ outTr,
                 float* __restrict__ outF32,
                 int mode)
{
    __shared__ _Float16 sB[2][64][40];   // 32-k strip, +8 halves row pad

    const int tid    = threadIdx.x;
    const int lane   = tid & 31;
    const int wv     = tid >> 5;
    const int laneLo = lane & 15;
    const int half   = lane >> 4;
    const int m0     = blockIdx.x * 256 + wv * 32;   // wave: 32 rows
    const int n0     = blockIdx.y * 64;

    // cooperative B-strip fetch: 64 rows x 32 halves, one b128 per thread
    const int r = tid >> 2;
    const int c = (tid & 3) * 8;
    const _Float16* gB = W + (size_t)(n0 + r) * HDIM + c;

    v8f acc[2][4] = {};

    const _Float16* aRow0 = A + (size_t)(m0 + laneLo) * HDIM + half * 16;
    const _Float16* aRow1 = aRow0 + (size_t)16 * HDIM;

    // preload strip 0
#if ASYNC_LDS
    __builtin_amdgcn_global_load_async_to_lds_b128(
        (gl_i4*)gB, (lds_i4*)&sB[0][r][c], 0, 0);
    WAIT_ASYNC();
#else
    *(v8h*)&sB[0][r][c] = *(const v8h*)gB;
#endif
    __syncthreads();

    for (int k0 = 0; k0 < HDIM; k0 += 32) {
        const int  buf  = (k0 >> 5) & 1;
        const bool more = (k0 + 32) < HDIM;

        // issue next strip early
#if ASYNC_LDS
        if (more)
            __builtin_amdgcn_global_load_async_to_lds_b128(
                (gl_i4*)(gB + k0 + 32), (lds_i4*)&sB[buf ^ 1][r][c], 0, 0);
#else
        v8h pre;
        if (more) pre = *(const v8h*)(gB + k0 + 32);
#endif

        v16h a0 = load16h(aRow0 + k0, aRow0 + k0 + 8);
        v16h a1 = load16h(aRow1 + k0, aRow1 + k0 + 8);

#pragma unroll
        for (int t = 0; t < 4; ++t) {
            const _Float16* bp = &sB[buf][t * 16 + laneLo][half * 16];
            v16h b = load16h(bp, bp + 8);
            acc[0][t] = wmma_f16(a0, b, acc[0][t]);
            acc[1][t] = wmma_f16(a1, b, acc[1][t]);
        }

#if ASYNC_LDS
        if (more) WAIT_ASYNC();
#else
        if (more) *(v8h*)&sB[buf ^ 1][r][c] = pre;
#endif
        __syncthreads();
    }

    if (mode == 0) {
#pragma unroll
        for (int f = 0; f < 2; ++f)
#pragma unroll
            for (int t = 0; t < 4; ++t)
#pragma unroll
                for (int j = 0; j < 8; ++j)
                    outPlain[(size_t)(m0 + f * 16 + j + half * 8) * HDIM +
                             (n0 + t * 16 + laneLo)] = (_Float16)acc[f][t][j];
    } else if (mode == 1) {
        // per-head transposed store: contiguous over m within a lane -> packed B128
        const int bh = (m0 >> 10) * NH + blockIdx.y;
#pragma unroll
        for (int f = 0; f < 2; ++f)
#pragma unroll
            for (int t = 0; t < 4; ++t) {
                int d = t * 16 + laneLo;
                v8h pk;
#pragma unroll
                for (int j = 0; j < 8; ++j) pk[j] = (_Float16)acc[f][t][j];
                *(v8h*)(outTr + ((size_t)bh * DH + d) * MSEQ +
                        (m0 & (MSEQ - 1)) + f * 16 + half * 8) = pk;
            }
    } else {
#pragma unroll
        for (int f = 0; f < 2; ++f)
#pragma unroll
            for (int t = 0; t < 4; ++t)
#pragma unroll
                for (int j = 0; j < 8; ++j)
                    outF32[(size_t)(m0 + f * 16 + j + half * 8) * HDIM +
                           (n0 + t * 16 + laneLo)] = acc[f][t][j];
    }
}

// ---------- stage 3: S = K^T V per (b,head); Kt,Vt: [B*NH, DH, MSEQ] f16 ----------
// Output column-major: Scol[bh*4096 + n*64 + k] = S[k][n]
__global__ void __launch_bounds__(256)
kv_kernel(const _Float16* __restrict__ Kt,
          const _Float16* __restrict__ Vt,
          _Float16* __restrict__ Scol)
{
    const int bh     = blockIdx.x;
    const int tid    = threadIdx.x;
    const int lane   = tid & 31;
    const int wv     = tid >> 5;
    const int laneLo = lane & 15;
    const int half   = lane >> 4;
    const int d1t    = wv & 3;
    const int d2t0   = (wv >> 2) * 2;

    v8f acc[2] = {};

    const _Float16* aRow =
        Kt + ((size_t)bh * DH + d1t * 16 + laneLo) * MSEQ + half * 16;
    const _Float16* b0Row =
        Vt + ((size_t)bh * DH + d2t0 * 16 + laneLo) * MSEQ + half * 16;
    const _Float16* b1Row = b0Row + (size_t)16 * MSEQ;

    for (int k0 = 0; k0 < MSEQ; k0 += 32) {
        v16h a  = load16h(aRow + k0, aRow + k0 + 8);
        v16h b0 = load16h(b0Row + k0, b0Row + k0 + 8);
        v16h b1 = load16h(b1Row + k0, b1Row + k0 + 8);
        acc[0] = wmma_f16(a, b0, acc[0]);
        acc[1] = wmma_f16(a, b1, acc[1]);
    }

#pragma unroll
    for (int t = 0; t < 2; ++t) {
        int n = (d2t0 + t) * 16 + laneLo;
        v8h pk;
#pragma unroll
        for (int j = 0; j < 8; ++j) pk[j] = (_Float16)acc[t][j];
        *(v8h*)(Scol + (size_t)bh * (DH * DH) + n * DH + d1t * 16 + half * 8) = pk;
    }
}

// ---------- stage 4: O = Q @ S per head ----------
__global__ void __launch_bounds__(256)
qs_kernel(const _Float16* __restrict__ Q,
          const _Float16* __restrict__ Scol,
          _Float16* __restrict__ O)
{
    const int tid    = threadIdx.x;
    const int lane   = tid & 31;
    const int wv     = tid >> 5;
    const int laneLo = lane & 15;
    const int half   = lane >> 4;
    const int m0     = (blockIdx.x * 8 + wv) * 16;
    const int head   = blockIdx.y;
    const int bh     = (m0 >> 10) * NH + head;

    v8f acc[4] = {};

    const _Float16* aRow  = Q + (size_t)(m0 + laneLo) * HDIM + head * DH + half * 16;
    const _Float16* sBase = Scol + (size_t)bh * (DH * DH);

    for (int k0 = 0; k0 < DH; k0 += 32) {
        v16h a = load16h(aRow + k0, aRow + k0 + 8);
#pragma unroll
        for (int t = 0; t < 4; ++t) {
            const _Float16* bPtr = sBase + (t * 16 + laneLo) * DH + k0 + half * 16;
            v16h b = load16h(bPtr, bPtr + 8);
            acc[t] = wmma_f16(a, b, acc[t]);
        }
    }

#pragma unroll
    for (int t = 0; t < 4; ++t)
#pragma unroll
        for (int j = 0; j < 8; ++j)
            O[(size_t)(m0 + j + half * 8) * HDIM + head * DH + t * 16 + laneLo] =
                (_Float16)acc[t][j];
}

// ---------- launcher ----------

extern "C" void kernel_launch(void* const* d_in, const int* in_sizes, int n_in,
                              void* d_out, int out_size, void* d_ws, size_t ws_size,
                              hipStream_t stream) {
    const float* h  = (const float*)d_in[0];
    // d_in[1] = key_pe : dead code in the reference output, unused
    const float* Wq = (const float*)d_in[2];
    const float* Wk = (const float*)d_in[3];
    const float* Wv = (const float*)d_in[4];
    const float* Wo = (const float*)d_in[5];
    float* out = (float*)d_out;

    char* ws = (char*)d_ws;
    const size_t actSz = (size_t)BM * HDIM * sizeof(_Float16);
    const size_t wSz   = (size_t)HDIM * HDIM * sizeof(_Float16);
    _Float16* hh   = (_Float16*)ws;  ws += actSz;
    _Float16* wqh  = (_Float16*)ws;  ws += wSz;
    _Float16* wkh  = (_Float16*)ws;  ws += wSz;
    _Float16* wvh  = (_Float16*)ws;  ws += wSz;
    _Float16* woh  = (_Float16*)ws;  ws += wSz;
    _Float16* Qb   = (_Float16*)ws;  ws += actSz;
    _Float16* Ktb  = (_Float16*)ws;  ws += actSz;
    _Float16* Vtb  = (_Float16*)ws;  ws += actSz;
    _Float16* Scol = (_Float16*)ws;  ws += (size_t)BSZ * NH * DH * DH * sizeof(_Float16);
    _Float16* Ob   = (_Float16*)ws;  ws += actSz;

    f32_to_f16_kernel<<<512, 256, 0, stream>>>(h,  hh,  BM * HDIM);
    f32_to_f16_kernel<<<256, 256, 0, stream>>>(Wq, wqh, HDIM * HDIM);
    f32_to_f16_kernel<<<256, 256, 0, stream>>>(Wk, wkh, HDIM * HDIM);
    f32_to_f16_kernel<<<256, 256, 0, stream>>>(Wv, wvh, HDIM * HDIM);
    f32_to_f16_kernel<<<256, 256, 0, stream>>>(Wo, woh, HDIM * HDIM);

    dim3 gridBig(BM / 256, HDIM / 64);   // (32, 16)
    dim3 blk(256);

    gemm_wmma_kernel<<<gridBig, blk, 0, stream>>>(hh, wqh, Qb, nullptr, nullptr, 0);
    gemm_wmma_kernel<<<gridBig, blk, 0, stream>>>(hh, wkh, nullptr, Ktb, nullptr, 1);
    gemm_wmma_kernel<<<gridBig, blk, 0, stream>>>(hh, wvh, nullptr, Vtb, nullptr, 1);

    kv_kernel<<<BSZ * NH, blk, 0, stream>>>(Ktb, Vtb, Scol);

    dim3 gridQS(BM / 128, NH);           // (64, 16)
    qs_kernel<<<gridQS, blk, 0, stream>>>(Qb, Scol, Ob);

    gemm_wmma_kernel<<<gridBig, blk, 0, stream>>>(Ob, woh, nullptr, nullptr, out, 2);
}
